// CausalSelfAttention_40862318854863
// MI455X (gfx1250) — matrix operand order repaired and checked by reading, not compiled, not run on hardware
//
#include <hip/hip_runtime.h>
#include <hip/hip_bf16.h>

typedef __attribute__((ext_vector_type(16))) __bf16 v16bf;
typedef __attribute__((ext_vector_type(8)))  float  v8f;
typedef __attribute__((ext_vector_type(2)))  __bf16 bf2;
typedef __attribute__((ext_vector_type(4)))  int    v4i;

#define B_   4
#define T_   2048
#define D_   1024
#define H_   16
#define HD_  64
#define N3_  3072
#define BT_  8192
#define SCALE_ 0.125f
#define QSZ_ ((size_t)BT_ * (size_t)D_)   // elements per Q/K/V/AO buffer

#if defined(__AMDGCN__) && __has_builtin(__builtin_amdgcn_global_load_async_to_lds_b128) && __has_builtin(__builtin_amdgcn_s_wait_asynccnt)
#define USE_ASYNC_COPY 1
#else
#define USE_ASYNC_COPY 0
#endif

#if USE_ASYNC_COPY
__device__ __forceinline__ void async_copy16(void* lds_dst, const void* gsrc) {
  typedef __attribute__((address_space(1))) v4i* gp_t;   // global int4*
  typedef __attribute__((address_space(3))) v4i* lp_t;   // LDS int4*
  __builtin_amdgcn_global_load_async_to_lds_b128(
      (gp_t)(v4i*)const_cast<void*>(gsrc),
      (lp_t)(v4i*)lds_dst, 0, 0);
}
__device__ __forceinline__ void async_wait0() {
  __builtin_amdgcn_s_wait_asynccnt(0);
}
#else
__device__ __forceinline__ void async_wait0() {}
#endif

__device__ __forceinline__ __bf16 f2bf(float f) { return (__bf16)f; }

union AF { v16bf v; bf2 p[8]; };

// ---------------------------------------------------------------------------
// Kernel 1: QKV projection  (M=8192, N=3072, K=1024), fp32 in -> bf16 QKV out
// Double-buffered LDS; next tile staged in VGPRs during compute.
// ---------------------------------------------------------------------------
__global__ __launch_bounds__(256) void qkv_proj_kernel(
    const float* __restrict__ x, const float* __restrict__ Wqkv,
    const float* __restrict__ bqkv, __bf16* __restrict__ ws) {
  __shared__ __align__(16) __bf16 As[2][128][34];
  __shared__ __align__(16) __bf16 Bt[2][128][34];
  const int tid  = threadIdx.x;
  const int lane = tid & 31;
  const int w    = tid >> 5;
  const int half = lane >> 4;
  const int l15  = lane & 15;
  const int mblock = blockIdx.y * 128;
  const int nblock = blockIdx.x * 128;

  v8f acc[8] = {};
  float4 regA[4], regB[4];

  auto issue_loads = [&](int kc) {
    #pragma unroll
    for (int j = 0; j < 4; ++j) {
      int idx4 = tid * 4 + j;
      int r = idx4 >> 3, c = (idx4 & 7) * 4;
      regA[j] = *reinterpret_cast<const float4*>(
          &x[(size_t)(mblock + r) * D_ + kc + c]);
    }
    #pragma unroll
    for (int j = 0; j < 4; ++j) {
      int idx4 = tid * 4 + j;
      int r = idx4 >> 5, c = (idx4 & 31) * 4;
      regB[j] = *reinterpret_cast<const float4*>(
          &Wqkv[(size_t)(kc + r) * N3_ + nblock + c]);
    }
  };
  auto store_tile = [&](int buf) {
    #pragma unroll
    for (int j = 0; j < 4; ++j) {
      int idx4 = tid * 4 + j;
      int r = idx4 >> 3, c = (idx4 & 7) * 4;
      bf2* d = reinterpret_cast<bf2*>(&As[buf][r][c]);
      bf2 t0; t0[0] = f2bf(regA[j].x); t0[1] = f2bf(regA[j].y);
      bf2 t1; t1[0] = f2bf(regA[j].z); t1[1] = f2bf(regA[j].w);
      d[0] = t0; d[1] = t1;
    }
    #pragma unroll
    for (int j = 0; j < 4; ++j) {
      int idx4 = tid * 4 + j;
      int r = idx4 >> 5, c = (idx4 & 31) * 4;
      Bt[buf][c + 0][r] = f2bf(regB[j].x);
      Bt[buf][c + 1][r] = f2bf(regB[j].y);
      Bt[buf][c + 2][r] = f2bf(regB[j].z);
      Bt[buf][c + 3][r] = f2bf(regB[j].w);
    }
  };

  issue_loads(0);
  store_tile(0);
  const int NK = D_ / 32;
  for (int kt = 0; kt < NK; ++kt) {
    __syncthreads();                       // tile kt visible in buf kt&1
    const int buf = kt & 1;
    const bool more = (kt + 1 < NK);
    if (more) issue_loads((kt + 1) * 32);  // overlap with compute below

    AF a;
    const int m = w * 16 + l15;
    #pragma unroll
    for (int v = 0; v < 8; ++v) {
      int kb = (half << 4) + (v << 1);
      a.p[v] = *reinterpret_cast<const bf2*>(&As[buf][m][kb]);
    }
    AF bfr[2];
    #pragma unroll
    for (int v = 0; v < 8; ++v)
      bfr[0].p[v] = *reinterpret_cast<const bf2*>(&Bt[buf][l15][(half << 4) + (v << 1)]);
    #pragma unroll
    for (int nb = 0; nb < 8; ++nb) {
      if (nb + 1 < 8) {
        int n = (nb + 1) * 16 + l15;
        #pragma unroll
        for (int v = 0; v < 8; ++v)
          bfr[(nb + 1) & 1].p[v] =
              *reinterpret_cast<const bf2*>(&Bt[buf][n][(half << 4) + (v << 1)]);
      }
      acc[nb] = __builtin_amdgcn_wmma_f32_16x16x32_bf16(
          false, a.v, false, bfr[nb & 1].v, (short)0, acc[nb], false, false);
    }
    if (more) store_tile(buf ^ 1);
  }

  // Epilogue: bias + scatter into Q/K/V [B,H,T,HD] bf16 (Q scaled)
  #pragma unroll
  for (int nb = 0; nb < 8; ++nb) {
    int col   = nblock + nb * 16 + l15;
    int which = col >> 10;
    int dcol  = col & 1023;
    int h = dcol >> 6, hd = dcol & 63;
    float bias = bqkv[col];
    #pragma unroll
    for (int vr = 0; vr < 8; ++vr) {
      int row = mblock + w * 16 + vr + (half << 3);
      int b = row >> 11, t = row & 2047;
      float val = acc[nb][vr] + bias;
      if (which == 0) val *= SCALE_;
      size_t off = (size_t)which * QSZ_ +
                   ((size_t)(b * H_ + h) * T_ + t) * HD_ + hd;
      ws[off] = f2bf(val);
    }
  }
}

// ---------------------------------------------------------------------------
// Kernel 2: causal flash attention. Grid (T/128, B*H). 8 waves x 16 queries.
// K tiles copied via async global->LDS when available; V staged + transposed.
// ---------------------------------------------------------------------------
__global__ __launch_bounds__(256) void flash_attn_kernel(
    const __bf16* __restrict__ wsqkv, __bf16* __restrict__ ao) {
  __shared__ __align__(16) __bf16 Kl[2][32][72];     // [key][hd], 144B rows
  __shared__ __align__(16) __bf16 Vt[2][64][34];     // [hd][key]
  __shared__ __align__(16) __bf16 Pl[8][16][34];     // per-wave P staging
  const int tid  = threadIdx.x;
  const int lane = tid & 31;
  const int w    = tid >> 5;
  const int half = lane >> 4;
  const int l15  = lane & 15;
  const int bh     = blockIdx.y;
  const int qstart = blockIdx.x * 128;

  const __bf16* Q = wsqkv;
  const __bf16* K = wsqkv + QSZ_;
  const __bf16* V = wsqkv + 2 * QSZ_;

  // Q fragments (A layout, hd split 0..31 / 32..63)
  AF aq[2];
  {
    int mg = qstart + w * 16 + l15;
    size_t qrow = ((size_t)bh * T_ + mg) * HD_;
    #pragma unroll
    for (int hh = 0; hh < 2; ++hh)
      #pragma unroll
      for (int v = 0; v < 8; ++v) {
        int kb = hh * 32 + (half << 4) + (v << 1);
        aq[hh].p[v] = *reinterpret_cast<const bf2*>(&Q[qrow + kb]);
      }
  }

  const int cr = (tid * 8) >> 6;       // copy row (key)
  const int cc = (tid * 8) & 63;       // copy col (hd)
  uint4 vreg;
#if !USE_ASYNC_COPY
  uint4 kreg;
#endif
  auto issue_chunk = [&](int kcchunk, int buf) {
    size_t base = ((size_t)bh * T_ + kcchunk + cr) * HD_ + cc;
#if USE_ASYNC_COPY
    async_copy16(&Kl[buf][cr][cc], &K[base]);
#else
    kreg = *reinterpret_cast<const uint4*>(&K[base]);
#endif
    vreg = *reinterpret_cast<const uint4*>(&V[base]);
  };
  auto finish_chunk = [&](int buf) {
#if !USE_ASYNC_COPY
    *reinterpret_cast<uint4*>(&Kl[buf][cr][cc]) = kreg;
#endif
    union { uint4 u; __bf16 b[8]; } vv; vv.u = vreg;
    #pragma unroll
    for (int j = 0; j < 8; ++j) Vt[buf][cc + j][cr] = vv.b[j];
  };

  v8f accO[4] = {};
  float m_i[8], l_i[8];
  #pragma unroll
  for (int i = 0; i < 8; ++i) { m_i[i] = -1e30f; l_i[i] = 0.0f; }

  const int wave_qmax = qstart + w * 16 + 15;  // wave-uniform causal frontier
  const int nk = (qstart + 128) / 32;          // chunks for this block

  issue_chunk(0, 0);
  finish_chunk(0);
  async_wait0();
  __syncthreads();

  for (int kt = 0; kt < nk; ++kt) {
    const int buf = kt & 1;
    const int kc  = kt * 32;
    const bool more = (kt + 1 < nk);
    if (more) issue_chunk(kc + 32, buf ^ 1);

    if (kc <= wave_qmax) {            // uniform branch -> EXEC stays all ones
      v8f s0 = {}, s1 = {};
      #pragma unroll
      for (int hh = 0; hh < 2; ++hh) {
        AF b0, b1;
        #pragma unroll
        for (int v = 0; v < 8; ++v) {
          int kb = hh * 32 + (half << 4) + (v << 1);
          b0.p[v] = *reinterpret_cast<const bf2*>(&Kl[buf][l15][kb]);
          b1.p[v] = *reinterpret_cast<const bf2*>(&Kl[buf][16 + l15][kb]);
        }
        s0 = __builtin_amdgcn_wmma_f32_16x16x32_bf16(
            false, aq[hh].v, false, b0.v, (short)0, s0, false, false);
        s1 = __builtin_amdgcn_wmma_f32_16x16x32_bf16(
            false, aq[hh].v, false, b1.v, (short)0, s1, false, false);
      }

      // online softmax (row stats per 16-lane half, C-layout rows = vr+8*half)
      int rowg_base = qstart + w * 16 + (half << 3);
      #pragma unroll
      for (int vr = 0; vr < 8; ++vr) {
        int rowg = rowg_base + vr;
        int c0 = kc + l15, c1 = kc + 16 + l15;
        float v0 = (c0 <= rowg) ? s0[vr] : -1e30f;
        float v1 = (c1 <= rowg) ? s1[vr] : -1e30f;
        float rmax = fmaxf(v0, v1);
        #pragma unroll
        for (int off = 1; off < 16; off <<= 1)
          rmax = fmaxf(rmax, __shfl_xor(rmax, off, 32));
        float mnew = fmaxf(m_i[vr], rmax);
        float corr = __expf(m_i[vr] - mnew);
        float p0 = __expf(v0 - mnew);
        float p1 = __expf(v1 - mnew);
        float rs = p0 + p1;
        #pragma unroll
        for (int off = 1; off < 16; off <<= 1)
          rs += __shfl_xor(rs, off, 32);
        l_i[vr] = l_i[vr] * corr + rs;
        m_i[vr] = mnew;
        #pragma unroll
        for (int ot = 0; ot < 4; ++ot) accO[ot][vr] *= corr;
        int mrow = vr + (half << 3);
        Pl[w][mrow][l15]      = f2bf(p0);
        Pl[w][mrow][16 + l15] = f2bf(p1);
      }

      // O += P x V
      AF ap;
      #pragma unroll
      for (int v = 0; v < 8; ++v) {
        int kb = (half << 4) + (v << 1);
        ap.p[v] = *reinterpret_cast<const bf2*>(&Pl[w][l15][kb]);
      }
      #pragma unroll
      for (int ot = 0; ot < 4; ++ot) {
        AF bv;
        #pragma unroll
        for (int v = 0; v < 8; ++v)
          bv.p[v] = *reinterpret_cast<const bf2*>(
              &Vt[buf][ot * 16 + l15][(half << 4) + (v << 1)]);
        accO[ot] = __builtin_amdgcn_wmma_f32_16x16x32_bf16(
            false, ap.v, false, bv.v, (short)0, accO[ot], false, false);
      }
    }

    if (more) finish_chunk(buf ^ 1);
    async_wait0();
    __syncthreads();
  }

  // normalize and write AO as bf16 [B,T,D] (rows b*T+t, cols h*64+hd)
  int b = bh >> 4, h = bh & 15;
  #pragma unroll
  for (int vr = 0; vr < 8; ++vr) {
    float inv = 1.0f / l_i[vr];
    int qg = qstart + w * 16 + vr + (half << 3);
    #pragma unroll
    for (int ot = 0; ot < 4; ++ot) {
      int hd = ot * 16 + l15;
      ao[((size_t)(b * T_ + qg)) * D_ + h * HD_ + hd] = f2bf(accO[ot][vr] * inv);
    }
  }
}

// ---------------------------------------------------------------------------
// Kernel 3: output projection  (M=8192, N=1024, K=1024), bf16 A, fp32 out
// A tile copied async (no conversion needed); B converted via staged regs.
// ---------------------------------------------------------------------------
__global__ __launch_bounds__(256) void out_proj_kernel(
    const __bf16* __restrict__ ao, const float* __restrict__ Wout,
    const float* __restrict__ bout, float* __restrict__ out) {
  __shared__ __align__(16) __bf16 As[2][128][40];    // 80B rows (16B aligned)
  __shared__ __align__(16) __bf16 Bt[2][128][34];
  const int tid  = threadIdx.x;
  const int lane = tid & 31;
  const int w    = tid >> 5;
  const int half = lane >> 4;
  const int l15  = lane & 15;
  const int mblock = blockIdx.y * 128;
  const int nblock = blockIdx.x * 128;

  v8f acc[8] = {};
  float4 regB[4];
#if !USE_ASYNC_COPY
  uint4 regA[2];
#endif

  auto issue_loads = [&](int kc, int buf) {
    #pragma unroll
    for (int j = 0; j < 2; ++j) {
      int idx8 = tid * 2 + j;
      int r = idx8 >> 2, c = (idx8 & 3) * 8;
      const void* src = &ao[(size_t)(mblock + r) * D_ + kc + c];
#if USE_ASYNC_COPY
      async_copy16(&As[buf][r][c], src);
#else
      regA[j] = *reinterpret_cast<const uint4*>(src);
#endif
    }
    #pragma unroll
    for (int j = 0; j < 4; ++j) {
      int idx4 = tid * 4 + j;
      int r = idx4 >> 5, c = (idx4 & 31) * 4;
      regB[j] = *reinterpret_cast<const float4*>(
          &Wout[(size_t)(kc + r) * D_ + nblock + c]);
    }
  };
  auto store_tile = [&](int buf) {
#if !USE_ASYNC_COPY
    #pragma unroll
    for (int j = 0; j < 2; ++j) {
      int idx8 = tid * 2 + j;
      int r = idx8 >> 2, c = (idx8 & 3) * 8;
      *reinterpret_cast<uint4*>(&As[buf][r][c]) = regA[j];
    }
#endif
    #pragma unroll
    for (int j = 0; j < 4; ++j) {
      int idx4 = tid * 4 + j;
      int r = idx4 >> 5, c = (idx4 & 31) * 4;
      Bt[buf][c + 0][r] = f2bf(regB[j].x);
      Bt[buf][c + 1][r] = f2bf(regB[j].y);
      Bt[buf][c + 2][r] = f2bf(regB[j].z);
      Bt[buf][c + 3][r] = f2bf(regB[j].w);
    }
  };

  issue_loads(0, 0);
  store_tile(0);
  async_wait0();
  const int NK = D_ / 32;
  for (int kt = 0; kt < NK; ++kt) {
    __syncthreads();
    const int buf = kt & 1;
    const bool more = (kt + 1 < NK);
    if (more) issue_loads((kt + 1) * 32, buf ^ 1);

    AF a;
    const int m = w * 16 + l15;
    #pragma unroll
    for (int v = 0; v < 8; ++v) {
      int kb = (half << 4) + (v << 1);
      a.p[v] = *reinterpret_cast<const bf2*>(&As[buf][m][kb]);
    }
    AF bfr[2];
    #pragma unroll
    for (int v = 0; v < 8; ++v)
      bfr[0].p[v] = *reinterpret_cast<const bf2*>(&Bt[buf][l15][(half << 4) + (v << 1)]);
    #pragma unroll
    for (int nb = 0; nb < 8; ++nb) {
      if (nb + 1 < 8) {
        int n = (nb + 1) * 16 + l15;
        #pragma unroll
        for (int v = 0; v < 8; ++v)
          bfr[(nb + 1) & 1].p[v] =
              *reinterpret_cast<const bf2*>(&Bt[buf][n][(half << 4) + (v << 1)]);
      }
      acc[nb] = __builtin_amdgcn_wmma_f32_16x16x32_bf16(
          false, a.v, false, bfr[nb & 1].v, (short)0, acc[nb], false, false);
    }
    if (more) store_tile(buf ^ 1);
    async_wait0();
  }

  #pragma unroll
  for (int nb = 0; nb < 8; ++nb) {
    int col = nblock + nb * 16 + l15;
    float bias = bout[col];
    #pragma unroll
    for (int vr = 0; vr < 8; ++vr) {
      int row = mblock + w * 16 + vr + (half << 3);
      out[(size_t)row * D_ + col] = acc[nb][vr] + bias;
    }
  }
}

// ---------------------------------------------------------------------------
extern "C" void kernel_launch(void* const* d_in, const int* in_sizes, int n_in,
                              void* d_out, int out_size, void* d_ws, size_t ws_size,
                              hipStream_t stream) {
  (void)in_sizes; (void)n_in; (void)out_size; (void)ws_size;
  const float* x    = (const float*)d_in[0];
  const float* Wqkv = (const float*)d_in[1];
  const float* bqkv = (const float*)d_in[2];
  const float* Wout = (const float*)d_in[3];
  const float* bout = (const float*)d_in[4];
  __bf16* ws  = (__bf16*)d_ws;          // Q | K | V | AO, each QSZ_ bf16
  __bf16* ao  = ws + 3 * QSZ_;
  float*  out = (float*)d_out;

  qkv_proj_kernel<<<dim3(N3_ / 128, BT_ / 128), 256, 0, stream>>>(x, Wqkv, bqkv, ws);
  flash_attn_kernel<<<dim3(T_ / 128, B_ * H_), 256, 0, stream>>>(ws, ao);
  out_proj_kernel<<<dim3(D_ / 128, BT_ / 128), 256, 0, stream>>>(ao, Wout, bout, out);
}
